// DeformableTransformer_22789096473140
// MI455X (gfx1250) — matrix operand order repaired and checked by reading, not compiled, not run on hardware
//
#include <hip/hip_runtime.h>
#include <stdint.h>

// ---------------- problem constants ----------------
#define BQ   2
#define LQn  13294
#define Dn   256
#define Hn   8
#define Ln   4
#define Pn   4
#define FFn  1024
#define DHn  32
#define MROWS (BQ * LQn)   // 26588

typedef __attribute__((ext_vector_type(16))) __bf16 v16bf;
typedef __attribute__((ext_vector_type(8)))  float  v8f;

union V16BF { uint4 u[2]; v16bf v; };

__device__ __host__ inline unsigned short f2bf(float f) {
  unsigned u = __builtin_bit_cast(unsigned, f);
  u += 0x7FFFu + ((u >> 16) & 1u);   // round-to-nearest-even
  return (unsigned short)(u >> 16);
}

// ---------------- f32 -> bf16 weight convert ----------------
__global__ void cvt_bf16_kernel(const float* __restrict__ x,
                                unsigned short* __restrict__ y, int n) {
  int i = blockIdx.x * 256 + threadIdx.x;
  if (i < n) y[i] = f2bf(x[i]);
}

// ---------------- src/pos -> src_bf, q_bf = bf16(src+pos) ----------------
__global__ void prep_kernel(const float4* __restrict__ src,
                            const float4* __restrict__ pos,
                            uint2* __restrict__ srcbf,
                            uint2* __restrict__ qbf, int n4) {
  int i = blockIdx.x * 256 + threadIdx.x;
  if (i >= n4) return;
  float4 s = src[i];
  float4 p = pos[i];
  uint2 a, q;
  a.x = (unsigned)f2bf(s.x) | ((unsigned)f2bf(s.y) << 16);
  a.y = (unsigned)f2bf(s.z) | ((unsigned)f2bf(s.w) << 16);
  q.x = (unsigned)f2bf(s.x + p.x) | ((unsigned)f2bf(s.y + p.y) << 16);
  q.y = (unsigned)f2bf(s.z + p.z) | ((unsigned)f2bf(s.w + p.w) << 16);
  srcbf[i] = a;
  qbf[i] = q;
}

// ---------------- generic bf16 GEMM: C[M,N] = A[M,K] * W[N,K]^T + bias ----
// One wave computes a 32(M) x 64(N) tile: 2 A fragments x 4 B fragments =
// 8 v_wmma_f32_16x16x32_bf16 per K-step; every B fragment is reused across
// two M-tiles (12 b128 loads -> 8 WMMAs per iteration).
#define GEMM_WAVES 4
__global__ __launch_bounds__(128) void gemm_bf16_wmma(
    const unsigned short* __restrict__ A,  // [M,K] bf16
    const unsigned short* __restrict__ W,  // [N,K] bf16
    const float* __restrict__ bias,        // [N]
    float* __restrict__ Cf,                // [M,N] f32 (or null)
    unsigned short* __restrict__ Cbf,      // [M,N] bf16 (or null)
    int M, int N, int K, int relu) {
  int lane = threadIdx.x & 31;
  int wave = threadIdx.x >> 5;
  int tile_m = blockIdx.x * GEMM_WAVES + wave;
  int m0 = tile_m * 32;
  if (m0 >= M) return;                 // wave-uniform: EXEC stays all-ones
  int n0 = blockIdx.y * 64;
  int half = lane >> 4;                // 0: lanes 0-15, 1: lanes 16-31
  int l15 = lane & 15;

  int arow0 = m0 + l15;       if (arow0 > M - 1) arow0 = M - 1;  // clamp pad rows
  int arow1 = m0 + 16 + l15;  if (arow1 > M - 1) arow1 = M - 1;
  const unsigned short* Arow0 = A + (size_t)arow0 * K;
  const unsigned short* Arow1 = A + (size_t)arow1 * K;

  v8f acc[8] = {};   // [mt*4 + nt]
  for (int ks = 0; ks < K; ks += 32) {
    // A fragment (16x32 bf16): lanes<16 hold K {0-7,16-23}, lanes>=16 {8-15,24-31}
    int ka = ks + (half ? 8 : 0);
    V16BF a0, a1;
    a0.u[0] = *(const uint4*)(Arow0 + ka);
    a0.u[1] = *(const uint4*)(Arow0 + ka + 16);
    a1.u[0] = *(const uint4*)(Arow1 + ka);
    a1.u[1] = *(const uint4*)(Arow1 + ka + 16);
    // B fragment (32x16 bf16): lanes<16 hold K {0-7,16-23}, lanes>=16 {8-15,24-31}
    int kb = ks + (half ? 8 : 0);
#pragma unroll
    for (int nt = 0; nt < 4; ++nt) {
      const unsigned short* Wr = W + (size_t)(n0 + nt * 16 + l15) * K + kb;
      V16BF b;
      b.u[0] = *(const uint4*)(Wr);
      b.u[1] = *(const uint4*)(Wr + 16);
      acc[nt]     = __builtin_amdgcn_wmma_f32_16x16x32_bf16(false, a0.v, false, b.v, (short)0, acc[nt],     false, false);
      acc[4 + nt] = __builtin_amdgcn_wmma_f32_16x16x32_bf16(false, a1.v, false, b.v, (short)0, acc[4 + nt], false, false);
    }
  }

  // C/D layout: lane l15 = column, half selects M rows 0-7 / 8-15, VGPR v = row
#pragma unroll
  for (int mt = 0; mt < 2; ++mt) {
    int mrow0 = m0 + mt * 16 + half * 8;
#pragma unroll
    for (int nt = 0; nt < 4; ++nt) {
      int c = n0 + nt * 16 + l15;
      float bv = bias ? bias[c] : 0.f;
      v8f av = acc[mt * 4 + nt];
#pragma unroll
      for (int v = 0; v < 8; ++v) {
        int m = mrow0 + v;
        if (m < M) {
          float val = av[v] + bv;
          if (relu) val = val > 0.f ? val : 0.f;
          if (Cf)  Cf[(size_t)m * N + c] = val;
          if (Cbf) Cbf[(size_t)m * N + c] = f2bf(val);
        }
      }
    }
  }
}

// ---------------- fused softmax + multi-scale deformable sampling ----------
// one thread per (token, head); 32-dim fp32 accumulator, bf16 output row.
__global__ __launch_bounds__(256) void msdeform_kernel(
    const float* __restrict__ awraw,   // [M,128]  h*16 + l*4+p
    const float* __restrict__ offb,    // [M,256]  h*32 + l*8 + p*2 + c
    const float* __restrict__ refp,    // [B,LQ,4,2]
    const float* __restrict__ value,   // [M,256]  h*32 + dh
    unsigned short* __restrict__ attnbf) {  // [M,256] bf16
  int t = blockIdx.x * 256 + threadIdx.x;
  if (t >= MROWS * Hn) return;
  int m = t >> 3;
  int h = t & 7;
  int b = m / LQn;

  // softmax over the 16 (level,point) logits of this head
  float w[16];
  const float* ap = awraw + (size_t)m * 128 + h * 16;
  float mx = -1e30f;
#pragma unroll
  for (int j = 0; j < 16; ++j) { w[j] = ap[j]; mx = fmaxf(mx, w[j]); }
  float s = 0.f;
#pragma unroll
  for (int j = 0; j < 16; ++j) { w[j] = __expf(w[j] - mx); s += w[j]; }
  float inv = 1.f / s;
#pragma unroll
  for (int j = 0; j < 16; ++j) w[j] *= inv;

  const float* op = offb + (size_t)m * 256 + h * 32;
  float4 acc[8];
#pragma unroll
  for (int j = 0; j < 8; ++j) acc[j] = make_float4(0.f, 0.f, 0.f, 0.f);

  const int HWs[4] = {100, 50, 25, 13};
  const int S0s[4] = {0, 10000, 12500, 13125};

  for (int lvl = 0; lvl < 4; ++lvl) {
    int HW = HWs[lvl];
    float fHW = (float)HW;
    int s0 = S0s[lvl];
    float rx = refp[((size_t)m * 4 + lvl) * 2 + 0];
    float ry = refp[((size_t)m * 4 + lvl) * 2 + 1];
    for (int p = 0; p < 4; ++p) {
      float ox = op[lvl * 8 + p * 2 + 0];
      float oy = op[lvl * 8 + p * 2 + 1];
      // grid_sample align_corners=False, zero padding; normalizer=(W,H)=HW
      float x = (rx + ox / fHW) * fHW - 0.5f;
      float y = (ry + oy / fHW) * fHW - 0.5f;
      float x0f = floorf(x), y0f = floorf(y);
      float wx1 = x - x0f, wx0 = 1.f - wx1;
      float wy1 = y - y0f, wy0 = 1.f - wy1;
      int x0 = (int)x0f, y0 = (int)y0f;
      float aw = w[lvl * 4 + p];
      int   cxi[4] = {x0, x0 + 1, x0, x0 + 1};
      int   cyi[4] = {y0, y0, y0 + 1, y0 + 1};
      float cw[4]  = {aw * wx0 * wy0, aw * wx1 * wy0, aw * wx0 * wy1, aw * wx1 * wy1};
#pragma unroll
      for (int c = 0; c < 4; ++c) {
        int xi = cxi[c], yi = cyi[c];
        if (xi >= 0 && xi < HW && yi >= 0 && yi < HW) {
          float wc = cw[c];
          const float4* vp = (const float4*)(value +
              (size_t)(b * LQn + s0 + yi * HW + xi) * 256 + h * 32);
#pragma unroll
          for (int j = 0; j < 8; ++j) {
            float4 vv = vp[j];
            acc[j].x += wc * vv.x; acc[j].y += wc * vv.y;
            acc[j].z += wc * vv.z; acc[j].w += wc * vv.w;
          }
        }
      }
    }
  }

  unsigned short* outp = attnbf + (size_t)m * 256 + h * 32;
#pragma unroll
  for (int j = 0; j < 8; ++j) {
    outp[j * 4 + 0] = f2bf(acc[j].x);
    outp[j * 4 + 1] = f2bf(acc[j].y);
    outp[j * 4 + 2] = f2bf(acc[j].z);
    outp[j * 4 + 3] = f2bf(acc[j].w);
  }
}

// ---------------- residual + LayerNorm (256 dims), one wave32 per row ------
__global__ __launch_bounds__(256) void ln_residual_kernel(
    const float* __restrict__ a,     // delta
    const float* __restrict__ res,   // residual input
    const float* __restrict__ g, const float* __restrict__ be,
    float* __restrict__ xout,            // f32 out (or null)
    unsigned short* __restrict__ xbf,    // bf16 out (or null)
    int M) {
  int row = blockIdx.x * 8 + (threadIdx.x >> 5);
  if (row >= M) return;                 // wave-uniform
  int lane = threadIdx.x & 31;
  const float* pa = a + (size_t)row * 256;
  const float* pr = res + (size_t)row * 256;
  float v[8];
  float s = 0.f;
#pragma unroll
  for (int j = 0; j < 8; ++j) { v[j] = pa[lane + j * 32] + pr[lane + j * 32]; s += v[j]; }
#pragma unroll
  for (int o = 16; o > 0; o >>= 1) s += __shfl_xor(s, o, 32);
  float mu = s * (1.f / 256.f);
  float var = 0.f;
#pragma unroll
  for (int j = 0; j < 8; ++j) { float d = v[j] - mu; var += d * d; }
#pragma unroll
  for (int o = 16; o > 0; o >>= 1) var += __shfl_xor(var, o, 32);
  var *= (1.f / 256.f);
  float rstd = rsqrtf(var + 1e-5f);
#pragma unroll
  for (int j = 0; j < 8; ++j) {
    int c = lane + j * 32;
    float out = (v[j] - mu) * rstd * g[c] + be[c];
    if (xout) xout[(size_t)row * 256 + c] = out;
    if (xbf)  xbf[(size_t)row * 256 + c] = f2bf(out);
  }
}

// ---------------- host-side orchestration ----------------
extern "C" void kernel_launch(void* const* d_in, const int* in_sizes, int n_in,
                              void* d_out, int out_size, void* d_ws, size_t ws_size,
                              hipStream_t stream) {
  const float* src    = (const float*)d_in[0];
  const float* pos    = (const float*)d_in[1];
  const float* refp   = (const float*)d_in[2];
  const float* W_off  = (const float*)d_in[3];
  const float* b_off  = (const float*)d_in[4];
  const float* W_attn = (const float*)d_in[5];
  const float* b_attn = (const float*)d_in[6];
  const float* W_val  = (const float*)d_in[7];
  const float* b_val  = (const float*)d_in[8];
  const float* W_out  = (const float*)d_in[9];
  const float* b_out  = (const float*)d_in[10];
  const float* W_ff1  = (const float*)d_in[11];
  const float* b_ff1  = (const float*)d_in[12];
  const float* W_ff2  = (const float*)d_in[13];
  const float* b_ff2  = (const float*)d_in[14];
  const float* g1     = (const float*)d_in[15];
  const float* be1    = (const float*)d_in[16];
  const float* g2     = (const float*)d_in[17];
  const float* be2    = (const float*)d_in[18];

  const int M = MROWS;
  char* w = (char*)d_ws;
  size_t o = 0;
  auto take = [&](size_t bytes) -> void* {
    void* p = (void*)(w + o);
    o = (o + bytes + 255) & ~(size_t)255;
    return p;
  };
  unsigned short* srcbf  = (unsigned short*)take((size_t)M * 256 * 2);
  unsigned short* qbf    = (unsigned short*)take((size_t)M * 256 * 2);
  float*          value  = (float*)take((size_t)M * 256 * 4);  // later: x (f32)
  float*          offb   = (float*)take((size_t)M * 256 * 4);  // later: y1 (out-proj)
  float*          awb    = (float*)take((size_t)M * 128 * 4);  // later: x_bf (bf16)
  unsigned short* attnbf = (unsigned short*)take((size_t)M * 256 * 2);
  unsigned short* hbf    = (unsigned short*)take((size_t)M * 1024 * 2);
  float*          y2     = (float*)take((size_t)M * 256 * 4);
  unsigned short* wv_bf   = (unsigned short*)take(256 * 256 * 2);
  unsigned short* woff_bf = (unsigned short*)take(256 * 256 * 2);
  unsigned short* wattn_bf= (unsigned short*)take(128 * 256 * 2);
  unsigned short* wout_bf = (unsigned short*)take(256 * 256 * 2);
  unsigned short* wff1_bf = (unsigned short*)take(1024 * 256 * 2);
  unsigned short* wff2_bf = (unsigned short*)take(1024 * 256 * 2);
  float* xf  = value;                    // aliases (ordering is safe on stream)
  float* y1  = offb;
  unsigned short* xbf = (unsigned short*)awb;

  // 1) convert weights to bf16
  auto cvt = [&](const float* in, unsigned short* out, int n) {
    cvt_bf16_kernel<<<(n + 255) / 256, 256, 0, stream>>>(in, out, n);
  };
  cvt(W_val, wv_bf, 256 * 256);
  cvt(W_off, woff_bf, 256 * 256);
  cvt(W_attn, wattn_bf, 128 * 256);
  cvt(W_out, wout_bf, 256 * 256);
  cvt(W_ff1, wff1_bf, 1024 * 256);
  cvt(W_ff2, wff2_bf, 1024 * 256);

  // 2) activations to bf16 (src, q = src+pos)
  int n4 = M * 256 / 4;
  prep_kernel<<<(n4 + 255) / 256, 256, 0, stream>>>(
      (const float4*)src, (const float4*)pos, (uint2*)srcbf, (uint2*)qbf, n4);

  int tilesM = (M + 31) / 32;            // 32 rows per wave tile
  dim3 gblk(128);
  dim3 ggrid((tilesM + GEMM_WAVES - 1) / GEMM_WAVES, 1);

  // 3) value = src @ W_val^T + b_val        [M,256]
  ggrid.y = 256 / 64;
  gemm_bf16_wmma<<<ggrid, gblk, 0, stream>>>(srcbf, wv_bf, b_val, value, nullptr, M, 256, 256, 0);
  // 4) off = q @ W_off^T + b_off            [M,256]
  gemm_bf16_wmma<<<ggrid, gblk, 0, stream>>>(qbf, woff_bf, b_off, offb, nullptr, M, 256, 256, 0);
  // 5) aw_raw = q @ W_attn^T + b_attn       [M,128]
  ggrid.y = 128 / 64;
  gemm_bf16_wmma<<<ggrid, gblk, 0, stream>>>(qbf, wattn_bf, b_attn, awb, nullptr, M, 128, 256, 0);

  // 6) softmax + deformable sampling -> attnbf (bf16)
  int nth = M * Hn;
  msdeform_kernel<<<(nth + 255) / 256, 256, 0, stream>>>(awb, offb, refp, value, attnbf);

  // 7) y1 = attn @ W_out^T + b_out          [M,256] (reuses offb)
  ggrid.y = 256 / 64;
  gemm_bf16_wmma<<<ggrid, gblk, 0, stream>>>(attnbf, wout_bf, b_out, y1, nullptr, M, 256, 256, 0);

  // 8) x = LN1(src + y1)  -> xf (f32, reuses value) and xbf (bf16, reuses awb)
  ln_residual_kernel<<<(M + 7) / 8, 256, 0, stream>>>(y1, src, g1, be1, xf, xbf, M);

  // 9) h = relu(x @ W_ff1^T + b_ff1) -> bf16 [M,1024]
  ggrid.y = 1024 / 64;
  gemm_bf16_wmma<<<ggrid, gblk, 0, stream>>>(xbf, wff1_bf, b_ff1, nullptr, hbf, M, 1024, 256, 1);

  // 10) y2 = h @ W_ff2^T + b_ff2            [M,256]
  ggrid.y = 256 / 64;
  gemm_bf16_wmma<<<ggrid, gblk, 0, stream>>>(hbf, wff2_bf, b_ff2, y2, nullptr, M, 256, 1024, 0);

  // 11) out = LN2(x + y2)
  ln_residual_kernel<<<(M + 7) / 8, 256, 0, stream>>>(y2, xf, g2, be2, (float*)d_out, nullptr, M);

  (void)in_sizes; (void)n_in; (void)out_size; (void)ws_size;
}
